// DeepseekMoE_42262478192987
// MI455X (gfx1250) — matrix operand order repaired and checked by reading, not compiled, not run on hardware
//
#include <hip/hip_runtime.h>
#include <math.h>

// Problem constants (from reference)
#define TT   512          // tokens
#define HH   2048         // hidden
#define II   1024         // moe intermediate
#define EE   32           // routed experts
#define KTOP 8            // experts per token
#define NGRP 8            // routing groups
#define TGRP 4            // top-k groups
#define NSHI 2048         // NSH * I (shared intermediate)
#define RSCALE 2.5f

// LDS B-stage geometry: 32 K-rows x 128 N-cols fp32, row stride 132 floats
// (528 B, 16B aligned). Rows 16..31 are additionally skewed by +16 floats so the
// two half-wave fragment-read groups (K and K+16) land on disjoint LDS banks.
#define BUFW  132
#define BUFSZ (32 * BUFW + 16)

typedef __bf16 v16bf __attribute__((ext_vector_type(16)));
typedef float  v8f   __attribute__((ext_vector_type(8)));

#if defined(__HIP_DEVICE_COMPILE__) && \
    __has_builtin(__builtin_amdgcn_global_load_async_to_lds_b128) && \
    __has_builtin(__builtin_amdgcn_s_wait_asynccnt)
#define USE_ASYNC 1
// exact param type per hipcc diagnostic: 'int __vector_size__(4*sizeof(int)) __device__ *'
typedef int v4i __attribute__((vector_size(4 * sizeof(int))));
typedef __attribute__((address_space(1))) v4i g_v4i;   // global src
typedef __attribute__((address_space(3))) v4i l_v4i;   // LDS dst
#else
#define USE_ASYNC 0
#endif

union AFrag { v16bf v; uint4 q[2]; };

// A fragment, 16x32 bf16 row-major. lane<16: M=lane, K={0..15};
// lane>=16: M=lane-16, K={16..31}. rowptr = &A[row][0].
__device__ __forceinline__ v16bf load_a_bf16(const __bf16* rowptr, int k0, bool hi) {
    AFrag f;
    const uint4* p = (const uint4*)(rowptr + k0 + (hi ? 16 : 0));
    f.q[0] = p[0];
    f.q[1] = p[1];
    return f.v;
}

__device__ __forceinline__ float silu_fast(float x) {
    // x * sigmoid(x) with v_rcp_f32 instead of the IEEE divide expansion
    return x * __builtin_amdgcn_rcpf(1.f + __expf(-x));
}

// 16-byte global -> LDS copy (async on CDNA5, sync fallback otherwise)
__device__ __forceinline__ void copy16_to_lds(const float* gp, float* lp) {
#if USE_ASYNC
    __builtin_amdgcn_global_load_async_to_lds_b128(
        (g_v4i*)(uintptr_t)gp,
        (l_v4i*)(unsigned)(uintptr_t)lp, 0, 0);
#else
    *(float4*)lp = *(const float4*)gp;
#endif
}

__device__ __forceinline__ void wait_async_all() {
#if USE_ASYNC
    __builtin_amdgcn_s_wait_asynccnt(0);
#endif
}

// Stage one 32 x 128 fp32 B panel into LDS, coalesced 16B per lane.
// Local col c maps to global col: c<64 -> gcb0+c, else gcb1+(c-64).
__device__ __forceinline__ void stage_b(const float* __restrict__ W, long ldw,
                                        int kbase, int gcb0, int gcb1,
                                        float* __restrict__ buf) {
    int tid = threadIdx.x;
    #pragma unroll
    for (int j = 0; j < 4; ++j) {
        int chunk = tid + j * 256;           // 1024 chunks = 32 rows * 32 chunks
        int row = chunk >> 5;
        int cf  = (chunk & 31) * 4;
        int gcol = (cf < 64) ? (gcb0 + cf) : (gcb1 + (cf - 64));
        const float* gp = W + (long)(kbase + row) * ldw + gcol;
        float* lp = buf + row * BUFW + cf + (row & 16);   // bank skew, keeps 16B align
        copy16_to_lds(gp, lp);
    }
}

// Double-buffered K-loop: async-stage B panel for k+1 while WMMA consumes k.
// lcol: this wave's 16-col tile base within the 128-wide panel (0..112).
__device__ __forceinline__ v8f wmma_gemm_staged(const __bf16* __restrict__ arow,
                                                const float* __restrict__ W, long ldw,
                                                int K, int gcb0, int gcb1, int lcol,
                                                float* __restrict__ bstage) {
    const int lane = threadIdx.x & 31;
    const bool hi = lane >= 16;
    const int kb2 = hi ? 16 : 0;
    const int l = lane & 15;
    v8f c = {};
    stage_b(W, ldw, 0, gcb0, gcb1, bstage);           // prologue into buf 0
    int cur = 0;
    for (int k0 = 0; k0 < K; k0 += 32) {
        wait_async_all();
        __syncthreads();                               // panel k0 ready for all waves
        if (k0 + 32 < K)
            stage_b(W, ldw, k0 + 32, gcb0, gcb1, bstage + (cur ^ 1) * BUFSZ);
        __builtin_prefetch(arow + k0 + 32, 0, 3);
        v16bf a = load_a_bf16(arow, k0, hi);
        const float* bb = bstage + cur * BUFSZ;
        v16bf b;
        #pragma unroll
        for (int i = 0; i < 16; ++i)
            b[i] = (__bf16)bb[(kb2 + i) * BUFW + kb2 + lcol + l];
        c = __builtin_amdgcn_wmma_f32_16x16x32_bf16(false, a, false, b, (short)0, c,
                                                    false, false);
        cur ^= 1;
    }
    return c;
}

// ---------------- kernel 0: zero per-expert counters ----------------
__global__ void k_init(int* __restrict__ counts) {
    if (threadIdx.x < EE) counts[threadIdx.x] = 0;
}

// ---------------- kernel 1: x fp32 -> bf16 ----------------
__global__ void k_cvt_bf16(const float* __restrict__ in, __bf16* __restrict__ out, int n4) {
    int i = blockIdx.x * blockDim.x + threadIdx.x;
    if (i >= n4) return;
    float4 v = ((const float4*)in)[i];
    union { __bf16 h[4]; uint2 u; } o;
    o.h[0] = (__bf16)v.x; o.h[1] = (__bf16)v.y;
    o.h[2] = (__bf16)v.z; o.h[3] = (__bf16)v.w;
    ((uint2*)out)[i] = o.u;
}

// ---------------- kernel 2: gating + grouped top-k routing ----------------
// One wave32 per token: lane e owns expert e.
__global__ __launch_bounds__(256)
void k_route(const float* __restrict__ x, const float* __restrict__ gw,
             const float* __restrict__ gb, float* __restrict__ cw,
             int* __restrict__ counts, int* __restrict__ tlist,
             float* __restrict__ wlist) {
    int wave = threadIdx.x >> 5;
    int lane = threadIdx.x & 31;
    int t = blockIdx.x * 8 + wave;
    if (t >= TT) return;

    const float4* xr = (const float4*)(x + (size_t)t * HH);
    const float4* wr = (const float4*)(gw + (size_t)lane * HH);
    float acc = 0.f;
    for (int k = 0; k < HH / 4; ++k) {
        float4 a = xr[k], b = wr[k];
        acc += a.x * b.x + a.y * b.y + a.z * b.z + a.w * b.w;
    }
    float s  = __builtin_amdgcn_rcpf(1.f + __expf(-acc));   // sigmoid
    float sc = s + gb[lane];

    // group score = sum of top-2 among the 4 experts of my group
    int g = lane >> 2;
    float a0 = __shfl(sc, g * 4 + 0), a1 = __shfl(sc, g * 4 + 1);
    float a2 = __shfl(sc, g * 4 + 2), a3 = __shfl(sc, g * 4 + 3);
    float M1 = fmaxf(a0, a1), n1 = fminf(a0, a1);
    float M2 = fmaxf(a2, a3), n2 = fminf(a2, a3);
    float top = fmaxf(M1, M2);
    float sec = (M1 >= M2) ? fmaxf(M2, n1) : fmaxf(M1, n2);
    float gscore = top + sec;

    int grank = 0;
    #pragma unroll
    for (int j = 0; j < NGRP; ++j) {
        float gj = __shfl(gscore, j * 4);
        grank += (gj > gscore) || (gj == gscore && j < g);
    }
    float msc = (grank < TGRP) ? sc : -1e30f;

    int r = 0;
    #pragma unroll
    for (int j = 0; j < EE; ++j) {
        float mj = __shfl(msc, j);
        r += (mj > msc) || (mj == msc && j < lane);
    }
    bool sel = (r < KTOP);
    float w = sel ? s : 0.f;
    float sum = w;
    #pragma unroll
    for (int off = 16; off > 0; off >>= 1) sum += __shfl_xor(sum, off);
    float wn = w * __builtin_amdgcn_rcpf(sum) * RSCALE;

    cw[t * EE + lane] = wn;
    if (sel) {
        int slot = atomicAdd(&counts[lane], 1);
        tlist[lane * TT + slot] = t;
        wlist[lane * TT + slot] = wn;
    }
}

// ---------------- kernel 3: shared expert gate/up GEMM + silu ----------------
__global__ __launch_bounds__(256)
void k_shared_gateup(const __bf16* __restrict__ xb, const float* __restrict__ wgu,
                     __bf16* __restrict__ sact) {
    __shared__ float bstage[2 * BUFSZ];
    __shared__ float lds_g[16][68];
    __shared__ float lds_u[16][68];
    int m0 = blockIdx.x * 16;
    int wave = threadIdx.x >> 5, lane = threadIdx.x & 31;
    bool hi = lane >= 16; int l = lane & 15;
    bool isUp = wave >= 4;
    int lcol = (isUp ? 64 : 0) + (wave & 3) * 16;

    const __bf16* arow = xb + (size_t)(m0 + l) * HH;
    v8f c = wmma_gemm_staged(arow, wgu, 2 * NSHI, HH,
                             blockIdx.y * 64, NSHI + blockIdx.y * 64, lcol, bstage);

    float (*buf)[68] = isUp ? lds_u : lds_g;
    int cbase = (wave & 3) * 16, mb = hi ? 8 : 0;
    #pragma unroll
    for (int r = 0; r < 8; ++r) buf[mb + r][cbase + l] = c[r];
    __syncthreads();
    for (int idx = threadIdx.x; idx < 16 * 64; idx += 256) {
        int row = idx >> 6, cc = idx & 63;
        float av = silu_fast(lds_g[row][cc]) * lds_u[row][cc];
        sact[(size_t)(m0 + row) * NSHI + blockIdx.y * 64 + cc] = (__bf16)av;
    }
}

// ---------------- kernel 4: shared expert down GEMM, writes d_out ----------------
__global__ __launch_bounds__(256)
void k_shared_down(const __bf16* __restrict__ sact, const float* __restrict__ wd,
                   float* __restrict__ out) {
    __shared__ float bstage[2 * BUFSZ];
    int m0 = blockIdx.x * 16;
    int wave = threadIdx.x >> 5, lane = threadIdx.x & 31;
    bool hi = lane >= 16; int l = lane & 15;
    int c0 = blockIdx.y * 128;

    const __bf16* arow = sact + (size_t)(m0 + l) * NSHI;
    v8f c = wmma_gemm_staged(arow, wd, HH, NSHI, c0, c0 + 64, wave * 16, bstage);

    int mb = hi ? 8 : 0;
    #pragma unroll
    for (int r = 0; r < 8; ++r)
        out[(size_t)(m0 + mb + r) * HH + c0 + wave * 16 + l] = c[r];
}

// ---------------- kernel 5: routed gate/up gather-GEMM + silu*cw ----------------
__global__ __launch_bounds__(256)
void k_routed_gateup(const __bf16* __restrict__ xb, const float* __restrict__ wgu,
                     const int* __restrict__ counts, const int* __restrict__ tlist,
                     const float* __restrict__ wlist, __bf16* __restrict__ act) {
    __shared__ float bstage[2 * BUFSZ];
    __shared__ float lds_g[16][68];
    __shared__ float lds_u[16][68];
    int e = blockIdx.x >> 5;
    int m0 = (blockIdx.x & 31) * 16;
    int cnt = counts[e];
    if (m0 >= cnt) return;

    int wave = threadIdx.x >> 5, lane = threadIdx.x & 31;
    bool hi = lane >= 16; int l = lane & 15;
    bool isUp = wave >= 4;
    int lcol = (isUp ? 64 : 0) + (wave & 3) * 16;

    int slot = m0 + l;
    int tok = tlist[e * TT + (slot < cnt ? slot : m0)];   // clamp ragged tail
    const __bf16* arow = xb + (size_t)tok * HH;
    const float* W = wgu + (size_t)e * HH * (2 * II);

    v8f c = wmma_gemm_staged(arow, W, 2 * II, HH,
                             blockIdx.y * 64, II + blockIdx.y * 64, lcol, bstage);

    float (*buf)[68] = isUp ? lds_u : lds_g;
    int cbase = (wave & 3) * 16, mb = hi ? 8 : 0;
    #pragma unroll
    for (int r = 0; r < 8; ++r) buf[mb + r][cbase + l] = c[r];
    __syncthreads();
    for (int idx = threadIdx.x; idx < 16 * 64; idx += 256) {
        int row = idx >> 6, cc = idx & 63;
        int s2 = m0 + row;
        if (s2 >= cnt) continue;
        float av = silu_fast(lds_g[row][cc]) * lds_u[row][cc] * wlist[e * TT + s2];
        act[((size_t)e * TT + s2) * II + blockIdx.y * 64 + cc] = (__bf16)av;
    }
}

// ---------------- kernel 6: routed down GEMM, atomicAdd into d_out ----------------
__global__ __launch_bounds__(256)
void k_routed_down(const __bf16* __restrict__ act, const float* __restrict__ wd,
                   const int* __restrict__ counts, const int* __restrict__ tlist,
                   float* __restrict__ out) {
    __shared__ float bstage[2 * BUFSZ];
    int e = blockIdx.x >> 5;
    int m0 = (blockIdx.x & 31) * 16;
    int cnt = counts[e];
    if (m0 >= cnt) return;

    int wave = threadIdx.x >> 5, lane = threadIdx.x & 31;
    bool hi = lane >= 16; int l = lane & 15;
    int c0 = blockIdx.y * 128;

    int slot = m0 + l;
    const __bf16* arow = act + ((size_t)e * TT + (slot < cnt ? slot : m0)) * II;
    const float* W = wd + (size_t)e * II * HH;

    v8f c = wmma_gemm_staged(arow, W, HH, II, c0, c0 + 64, wave * 16, bstage);

    int mb = hi ? 8 : 0;
    #pragma unroll
    for (int r = 0; r < 8; ++r) {
        int s2 = m0 + mb + r;
        if (s2 < cnt) {
            int tok = tlist[e * TT + s2];
            atomicAdd(&out[(size_t)tok * HH + c0 + wave * 16 + l], c[r]);
        }
    }
}

// ---------------- launch ----------------
extern "C" void kernel_launch(void* const* d_in, const int* in_sizes, int n_in,
                              void* d_out, int out_size, void* d_ws, size_t ws_size,
                              hipStream_t stream) {
    const float* x    = (const float*)d_in[0];  // [T,H]
    const float* gw   = (const float*)d_in[1];  // [E,H]
    const float* gb   = (const float*)d_in[2];  // [E]
    const float* wgu  = (const float*)d_in[3];  // [E,H,2I]
    const float* wdn  = (const float*)d_in[4];  // [E,I,H]
    const float* swgu = (const float*)d_in[5];  // [H, 2*NSH*I]
    const float* swd  = (const float*)d_in[6];  // [NSH*I, H]
    float* out = (float*)d_out;                 // [T,H]

    size_t off = 0;
    auto take = [&](size_t bytes) {
        void* p = (char*)d_ws + off;
        off += (bytes + 255) & ~(size_t)255;
        return p;
    };
    float*  cw     = (float*)take((size_t)TT * EE * 4);
    int*    counts = (int*)take(EE * 4);
    int*    tlist  = (int*)take((size_t)EE * TT * 4);
    float*  wlist  = (float*)take((size_t)EE * TT * 4);
    __bf16* xb     = (__bf16*)take((size_t)TT * HH * 2);
    __bf16* actb   = (__bf16*)take((size_t)EE * TT * II * 2);
    __bf16* sact   = (__bf16*)take((size_t)TT * NSHI * 2);

    k_init<<<1, 32, 0, stream>>>(counts);
    k_cvt_bf16<<<(TT * HH / 4 + 255) / 256, 256, 0, stream>>>(x, xb, TT * HH / 4);
    k_route<<<TT / 8, 256, 0, stream>>>(x, gw, gb, cw, counts, tlist, wlist);
    // shared expert first: overwrites d_out, routed experts then atomically add
    k_shared_gateup<<<dim3(TT / 16, NSHI / 64), 256, 0, stream>>>(xb, swgu, sact);
    k_shared_down<<<dim3(TT / 16, HH / 128), 256, 0, stream>>>(sact, swd, out);
    k_routed_gateup<<<dim3(EE * (TT / 16), II / 64), 256, 0, stream>>>(
        xb, wgu, counts, tlist, wlist, actb);
    k_routed_down<<<dim3(EE * (TT / 16), HH / 128), 256, 0, stream>>>(
        actb, wdn, counts, tlist, out);
    (void)in_sizes; (void)n_in; (void)out_size; (void)ws_size;
}
